// Net_83391085019262
// MI455X (gfx1250) — compile-verified, matrix operand order checked by reading, not run on hardware
//
#include <hip/hip_runtime.h>
#include <hip/hip_bf16.h>

// ---------------------------------------------------------------------------
// SNN (snntorch-style Leaky, subtract reset) on MI455X / gfx1250
//   Phase 1: cur1 = x @ W1^T + b1      -> f32 WMMA GEMM (exact precision)
//   Phase 2: 256-step leaky-integrate-fire loop, one workgroup per batch row,
//            state held in registers, wave32 shuffle + LDS reduction for the
//            thin spk1 @ W2^T (N=2) contraction.
// ---------------------------------------------------------------------------

typedef __attribute__((ext_vector_type(2)))  float    v2f;
typedef __attribute__((ext_vector_type(8)))  float    v8f;
typedef __attribute__((ext_vector_type(16))) _Float16 v16h;

#define B_SZ   128
#define IN_N   128
#define HID    1003
#define HID_P  1008   // 63 tiles of 16
#define T_STP  256
#define OUT_N  2
#define BETA   0.95f
#define THR    1.0f

// ---------------------------------------------------------------------------
// Kernel 1: cur1[b][h] = sum_k x[b][k] * W1[h][k] + b1[h]
// One wave32 per 16x16 output tile. grid = (8 M-tiles, 63 N-tiles).
//
// Out-of-range hidden rows (h in [1003,1008)) are handled by CLAMPING the row
// index instead of masking the value: the resulting pad columns of cur1 are
// never read by the time-loop kernel, and clamping keeps every load
// unconditional so EXEC stays all-1s (no saveexec/branch per load).
// ---------------------------------------------------------------------------
__global__ __launch_bounds__(32) void snn_fc1_wmma(
    const float* __restrict__ x,    // [128][128]
    const float* __restrict__ W1,   // [1003][128]
    const float* __restrict__ b1,   // [1003]
    float* __restrict__ cur1)       // [128][1008] (ws)
{
    const int tileM = blockIdx.x;       // batch tile 0..7
    const int tileN = blockIdx.y;       // hidden tile 0..62
    const int lane  = threadIdx.x;      // 0..31
    const int m     = tileM * 16 + (lane & 15);          // batch row (A)
    const int n     = tileN * 16 + (lane & 15);          // hidden row (B)
    const int nclmp = (n < HID) ? n : (HID - 1);         // clamp, don't mask

    v8f acc = {};

#if __has_builtin(__builtin_amdgcn_wmma_f32_16x16x4_f32)
    // f32 path: A 16x4, B 4x16, K advances 4 per WMMA (32 WMMAs total).
    // Layout (ISA 7.12.2): lanes 0-15 hold K={0,1}, lanes 16-31 hold K={2,3}.
    const int khalf = (lane >> 4) * 2;  // 0 or 2
    const float* __restrict__ xrow = x  + m * IN_N + khalf;
    const float* __restrict__ wrow = W1 + nclmp * IN_N + khalf;
    #pragma unroll
    for (int k0 = 0; k0 < IN_N; k0 += 4) {
        const v2f a  = *(const v2f*)(xrow + k0);   // 8B-aligned -> b64 load
        const v2f bm = *(const v2f*)(wrow + k0);   // 8B-aligned -> b64 load
        acc = __builtin_amdgcn_wmma_f32_16x16x4_f32(
            /*neg_a=*/false, a, /*neg_b=*/false, bm,
            /*c_mod=*/(short)0, acc, /*reuse_a=*/false, /*reuse_b=*/false);
    }
#else
    // Fallback: codegen-confirmed f16 WMMA (K=32 per op, 4 ops).
    const int kbase = (lane >> 4) * 8;  // 0 or 8
    const float* __restrict__ xrow = x  + m * IN_N;
    const float* __restrict__ wrow = W1 + nclmp * IN_N;
    #pragma unroll
    for (int k0 = 0; k0 < IN_N; k0 += 32) {
        v16h a, bm;
        #pragma unroll
        for (int e = 0; e < 8; ++e) {
            a[e]      = (_Float16)xrow[k0 + kbase + e];
            a[e + 8]  = (_Float16)xrow[k0 + 16 + kbase + e];
            bm[e]     = (_Float16)wrow[k0 + kbase + e];
            bm[e + 8] = (_Float16)wrow[k0 + 16 + kbase + e];
        }
        acc = __builtin_amdgcn_wmma_f32_16x16x32_f16(
            false, a, false, bm, (short)0, acc, false, false);
    }
#endif

    // C/D layout: lane<16 -> VGPR r = row r, col lane; lane>=16 -> row 8+r.
    const float bias    = b1[nclmp];                 // junk only in pad cols
    const int   rowBase = tileM * 16 + ((lane < 16) ? 0 : 8);
    const int   col     = tileN * 16 + (lane & 15);
    #pragma unroll
    for (int r = 0; r < 8; ++r) {
        cur1[(rowBase + r) * HID_P + col] = acc[r] + bias;
    }
}

// ---------------------------------------------------------------------------
// Kernel 2: 256-step LIF recurrence. One block (1024 thr = 32 wave32) per
// batch element; thread tid owns hidden unit h = tid. mem1/cur1/W2 columns
// live in registers for the whole loop; per-step cross-thread work is only
// the 2-wide weighted popcount of spikes.
// ---------------------------------------------------------------------------
__global__ __launch_bounds__(1024) void snn_time_loop(
    const float* __restrict__ cur1,  // [128][1008]
    const float* __restrict__ W2,    // [2][1003]
    const float* __restrict__ b2,    // [2]
    float* __restrict__ out)         // spk_rec [256*128*2] ++ mem_rec [256*128*2]
{
    const int b    = blockIdx.x;       // batch row
    const int tid  = threadIdx.x;      // 0..1023
    const int lane = tid & 31;
    const int wave = tid >> 5;
    const bool active = (tid < HID);

    const float c1  = active ? cur1[b * HID_P + tid] : 0.0f;
    const float w20 = active ? W2[tid]       : 0.0f;   // W2[0][h]
    const float w21 = active ? W2[HID + tid] : 0.0f;   // W2[1][h]
    const float bb0 = b2[0];
    const float bb1 = b2[1];

    float mem1 = 0.0f;
    float mem2_0 = 0.0f, mem2_1 = 0.0f;   // only meaningful on (wave0, lane0)

    __shared__ float part0[32];
    __shared__ float part1[32];

    for (int t = 0; t < T_STP; ++t) {
        // Layer 1: reset from OLD membrane, then leak + drive.
        const float reset1 = (mem1 > THR) ? 1.0f : 0.0f;
        mem1 = BETA * mem1 + c1 - THR * reset1;
        const bool spk = active && (mem1 > THR);

        float s0 = spk ? w20 : 0.0f;
        float s1 = spk ? w21 : 0.0f;

        // wave32 tree reduction (ds/permlane on gfx1250)
        #pragma unroll
        for (int off = 16; off > 0; off >>= 1) {
            s0 += __shfl_down(s0, off, 32);
            s1 += __shfl_down(s1, off, 32);
        }
        if (lane == 0) { part0[wave] = s0; part1[wave] = s1; }
        __syncthreads();

        if (wave == 0) {
            float p0 = part0[lane];
            float p1 = part1[lane];
            #pragma unroll
            for (int off = 16; off > 0; off >>= 1) {
                p0 += __shfl_down(p0, off, 32);
                p1 += __shfl_down(p1, off, 32);
            }
            if (lane == 0) {
                const float cur2_0 = p0 + bb0;
                const float cur2_1 = p1 + bb1;
                const float r0 = (mem2_0 > THR) ? 1.0f : 0.0f;
                const float r1 = (mem2_1 > THR) ? 1.0f : 0.0f;
                mem2_0 = BETA * mem2_0 + cur2_0 - THR * r0;
                mem2_1 = BETA * mem2_1 + cur2_1 - THR * r1;
                const int base = (t * B_SZ + b) * OUT_N;
                out[base + 0] = (mem2_0 > THR) ? 1.0f : 0.0f;   // spk_rec
                out[base + 1] = (mem2_1 > THR) ? 1.0f : 0.0f;
                out[T_STP * B_SZ * OUT_N + base + 0] = mem2_0;  // mem_rec
                out[T_STP * B_SZ * OUT_N + base + 1] = mem2_1;
            }
        }
        __syncthreads();   // protect part0/part1 for next step
    }
}

// ---------------------------------------------------------------------------
extern "C" void kernel_launch(void* const* d_in, const int* in_sizes, int n_in,
                              void* d_out, int out_size, void* d_ws, size_t ws_size,
                              hipStream_t stream)
{
    const float* x  = (const float*)d_in[0];   // [128][128]
    const float* W1 = (const float*)d_in[1];   // [1003][128]
    const float* b1 = (const float*)d_in[2];   // [1003]
    const float* W2 = (const float*)d_in[3];   // [2][1003]
    const float* b2 = (const float*)d_in[4];   // [2]
    float* out  = (float*)d_out;               // 2 * 256*128*2 floats
    float* cur1 = (float*)d_ws;                // 128*1008 floats = 516 KB

    dim3 g1(B_SZ / 16, HID_P / 16);            // (8, 63) tiles
    snn_fc1_wmma<<<g1, 32, 0, stream>>>(x, W1, b1, cur1);

    snn_time_loop<<<B_SZ, 1024, 0, stream>>>(cur1, W2, b2, out);
}
